// TokenRoutedFFN_30021821399690
// MI455X (gfx1250) — matrix operand order repaired and checked by reading, not compiled, not run on hardware
//
#include <hip/hip_runtime.h>
#include <hip/hip_bf16.h>
#include <math.h>

typedef __attribute__((ext_vector_type(16))) _Float16 v16h;
typedef __attribute__((ext_vector_type(8)))  _Float16 v8h;
typedef __attribute__((ext_vector_type(8)))  float    v8f;
typedef __attribute__((ext_vector_type(4)))  float    v4f;
typedef __attribute__((ext_vector_type(4)))  int      v4i;

#define B_   4
#define S_   4096
#define D_   1024
#define DFF_ 4096
#define K_   1024
#define MT   256
#define NT   64
#define KT   64
#define LDT  72   // LDS tile row stride in halves (64 data + 8 pad)

#define GLOBAL_AS __attribute__((address_space(1)))
#define LDS_AS    __attribute__((address_space(3)))

typedef LDS_AS _Float16       lds_h;
typedef const LDS_AS _Float16 clds_h;

#if defined(__has_builtin)
#if __has_builtin(__builtin_amdgcn_global_load_async_to_lds_b128)
#define HAVE_ASYNC_LDS 1
#endif
#endif

// 16-byte global -> LDS copy (async on CDNA5)
__device__ __forceinline__ void copy16(const _Float16* g, lds_h* l) {
#if defined(HAVE_ASYNC_LDS)
    __builtin_amdgcn_global_load_async_to_lds_b128(
        (GLOBAL_AS v4i*)(size_t)g, (LDS_AS v4i*)l, 0, 0);
#else
    *(LDS_AS v8h*)l = *(const v8h*)g;
#endif
}

// Wait for the *previous* tile's copies. If a next tile (10 chunks/thread) was
// just issued, waiting <=10 retires everything older (in-order completion).
// If nothing was issued (final iteration), must drain to 0.
__device__ __forceinline__ void wait_copies(bool issued_next) {
#if defined(HAVE_ASYNC_LDS)
    if (issued_next) __builtin_amdgcn_s_wait_asynccnt(10);
    else             __builtin_amdgcn_s_wait_asynccnt(0);
#endif
}

// ---------------- Router: scores[b*S+s] = dot(x[b,s,:], rw) + rb ----------------
__global__ __launch_bounds__(256) void router_k(const float* __restrict__ x,
                                                const float* __restrict__ rw,
                                                const float* __restrict__ rb,
                                                float* __restrict__ scores) {
    int wave = (blockIdx.x * 256 + threadIdx.x) >> 5;
    int lane = threadIdx.x & 31;
    if (wave >= B_ * S_) return;
    const float* xr = x + (size_t)wave * D_;
    float sum = 0.f;
    for (int i = 0; i < D_; i += 128) {
        v4f xv = *(const v4f*)(xr + i + lane * 4);
        v4f wv = *(const v4f*)(rw + i + lane * 4);
        sum += xv[0]*wv[0] + xv[1]*wv[1] + xv[2]*wv[2] + xv[3]*wv[3];
    }
    for (int off = 16; off > 0; off >>= 1) sum += __shfl_xor(sum, off, 32);
    if (lane == 0) scores[wave] = sum + rb[0];
}

// ------------- Exact top-K by rank counting (deterministic, no atomics) -------------
__global__ __launch_bounds__(1024) void select_k(const float* __restrict__ scores,
                                                 int* __restrict__ sel_idx,
                                                 float* __restrict__ gates) {
    __shared__ float sc[S_];
    int b = blockIdx.x;
    const float* srow = scores + b * S_;
    for (int i = threadIdx.x; i < S_; i += 1024) sc[i] = srow[i];
    __syncthreads();
    for (int t = threadIdx.x; t < S_; t += 1024) {
        float si = sc[t];
        int rank = 0;
        for (int j = 0; j < S_; ++j) {
            float sj = sc[j];
            rank += (int)((sj > si) || (sj == si && j < t));
        }
        if (rank < K_) {
            sel_idx[b * K_ + rank] = t;
            gates[b * K_ + rank] = 1.f / (1.f + expf(-si));
        }
    }
}

// ------------- Transposing fp32 -> f16 convert: dst[n][k] = (f16)src[k][n] -------------
__global__ __launch_bounds__(256) void cvtT_k(const float* __restrict__ src,
                                              _Float16* __restrict__ dst,
                                              int K, int N) {
    __shared__ float t[32][33];
    int kb = blockIdx.y * 32, nb = blockIdx.x * 32;
    for (int i = threadIdx.y; i < 32; i += 8)
        t[i][threadIdx.x] = src[(size_t)(kb + i) * N + nb + threadIdx.x];
    __syncthreads();
    for (int i = threadIdx.y; i < 32; i += 8)
        dst[(size_t)(nb + i) * K + kb + threadIdx.x] = (_Float16)t[threadIdx.x][i];
}

// ---------------- Gather selected rows of x into f16 ----------------
__global__ __launch_bounds__(256) void gather_k(const float* __restrict__ x,
                                                const int* __restrict__ sel_idx,
                                                _Float16* __restrict__ selh) {
    int m = blockIdx.x;
    int b = m >> 10;
    int tok = sel_idx[m];
    const float* src = x + ((size_t)b * S_ + tok) * D_;
    _Float16* dst = selh + (size_t)m * D_;
    for (int i = threadIdx.x * 4; i < D_; i += 256 * 4) {
        v4f v = *(const v4f*)(src + i);
        dst[i + 0] = (_Float16)v[0];
        dst[i + 1] = (_Float16)v[1];
        dst[i + 2] = (_Float16)v[2];
        dst[i + 3] = (_Float16)v[3];
    }
}

// ---------------- Zero output ----------------
__global__ void zero_k(float* __restrict__ p, int n4) {
    int stride = gridDim.x * blockDim.x;
    for (int i = blockIdx.x * blockDim.x + threadIdx.x; i < n4; i += stride) {
        v4f z; z[0] = 0.f; z[1] = 0.f; z[2] = 0.f; z[3] = 0.f;
        *(v4f*)(p + (size_t)i * 4) = z;
    }
}

// ---------------- WMMA fragment helpers (layouts per ISA 7.12.2) ----------------
__device__ __forceinline__ v16h load_afrag(clds_h* At, int row, int half, int kk) {
    clds_h* ap = At + row * LDT + kk + half * 8;
    v8h lo = *(const LDS_AS v8h*)ap;
    v8h hi = *(const LDS_AS v8h*)(ap + 16);
    v16h f;
#pragma unroll
    for (int e = 0; e < 8; ++e) { f[e] = lo[e]; f[8 + e] = hi[e]; }
    return f;
}
__device__ __forceinline__ v16h load_bfrag(clds_h* BtT, int col, int half, int kk) {
    clds_h* bp = BtT + col * LDT + kk + half * 8;
    v8h lo = *(const LDS_AS v8h*)bp;
    v8h hi = *(const LDS_AS v8h*)(bp + 16);
    v16h f;
#pragma unroll
    for (int e = 0; e < 8; ++e) { f[e] = lo[e]; f[8 + e] = hi[e]; }
    return f;
}

// ---------------- Tile staging: A (MTxKT) + Bt (NTxKT), both row-contiguous ----------------
// 10 copy16 per thread per tile (A: 8, B: 2)
__device__ __forceinline__ void stage_tile(const _Float16* __restrict__ A, size_t lda, int m0, int k0,
                                           const _Float16* __restrict__ Bw, size_t ldb, int n0,
                                           lds_h* At, lds_h* Bl, int tid) {
#pragma unroll
    for (int j = 0; j < 8; ++j) {             // A: 256 rows x 8 chunks
        int c = tid + 256 * j;
        int r = c >> 3, cc = c & 7;
        copy16(&A[(size_t)(m0 + r) * lda + k0 + cc * 8], &At[r * LDT + cc * 8]);
    }
#pragma unroll
    for (int j = 0; j < 2; ++j) {             // B: 64 rows x 8 chunks
        int c = tid + 256 * j;
        int n = c >> 3, cc = c & 7;
        copy16(&Bw[(size_t)(n0 + n) * ldb + k0 + cc * 8], &Bl[n * LDT + cc * 8]);
    }
}

// ---------------- GEMM mainloop (double-buffered LDS, async staging) ----------------
__device__ __forceinline__ void gemm_mainloop(const _Float16* __restrict__ A, size_t lda,
                                              const _Float16* __restrict__ Bw, size_t ldb,
                                              int m0, int n0, int kdim,
                                              lds_h* At0, lds_h* At1,
                                              lds_h* Bl0, lds_h* Bl1,
                                              int tid, int wv, int lrow, int half,
                                              v8f acc[2][4]) {
    lds_h* Abuf[2] = { At0, At1 };
    lds_h* Bbuf[2] = { Bl0, Bl1 };
    const int NK = kdim / KT;
    stage_tile(A, lda, m0, 0, Bw, ldb, n0, Abuf[0], Bbuf[0], tid);
    for (int i = 0; i < NK; ++i) {
        int cur = i & 1, nxt = cur ^ 1;
        bool more = (i + 1 < NK);
        if (more)
            stage_tile(A, lda, m0, (i + 1) * KT, Bw, ldb, n0, Abuf[nxt], Bbuf[nxt], tid);
        wait_copies(more);      // current tile's copies complete (in-order)
        __syncthreads();        // all waves' copies visible
#pragma unroll
        for (int kk = 0; kk < KT; kk += 32) {
            v16h af0 = load_afrag(Abuf[cur], wv * 32 + lrow, half, kk);
            v16h af1 = load_afrag(Abuf[cur], wv * 32 + 16 + lrow, half, kk);
#pragma unroll
            for (int t = 0; t < 4; ++t) {
                v16h bf = load_bfrag(Bbuf[cur], t * 16 + lrow, half, kk);
                acc[0][t] = __builtin_amdgcn_wmma_f32_16x16x32_f16(
                    false, af0, false, bf, (short)0, acc[0][t], false, false);
                acc[1][t] = __builtin_amdgcn_wmma_f32_16x16x32_f16(
                    false, af1, false, bf, (short)0, acc[1][t], false, false);
            }
        }
        __syncthreads();        // reads done before this buffer is overwritten
    }
}

// ---------------- GEMM1: H = gelu(selh @ w1t^T + b1), f16 out ----------------
__global__ __launch_bounds__(256) void gemm1_k(const _Float16* __restrict__ A,    // [B_*K_][D_]
                                               const _Float16* __restrict__ BwT,  // [DFF_][D_] (transposed w1)
                                               const float* __restrict__ bias,    // [DFF_]
                                               _Float16* __restrict__ H) {        // [B_*K_][DFF_]
    __shared__ _Float16 At[2][MT * LDT];
    __shared__ _Float16 Bl[2][NT * LDT];
    const int tid  = threadIdx.x;
    const int n0   = blockIdx.x * NT;
    const int m0   = blockIdx.y * MT;
    const int lane = tid & 31;
    const int wv   = tid >> 5;
    const int lrow = lane & 15;
    const int half = lane >> 4;

    v8f acc[2][4];
#pragma unroll
    for (int s = 0; s < 2; ++s)
#pragma unroll
        for (int t = 0; t < 4; ++t)
#pragma unroll
            for (int e = 0; e < 8; ++e) acc[s][t][e] = 0.f;

    gemm_mainloop(A, D_, BwT, D_, m0, n0, D_,
                  (lds_h*)&At[0][0], (lds_h*)&At[1][0],
                  (lds_h*)&Bl[0][0], (lds_h*)&Bl[1][0],
                  tid, wv, lrow, half, acc);

#pragma unroll
    for (int s = 0; s < 2; ++s)
#pragma unroll
        for (int t = 0; t < 4; ++t) {
            int nc = n0 + t * 16 + lrow;
            float bv = bias[nc];
#pragma unroll
            for (int r = 0; r < 8; ++r) {
                int mr = m0 + wv * 32 + s * 16 + r + half * 8;
                float v = acc[s][t][r] + bv;
                v = 0.5f * v * (1.0f + erff(v * 0.70710678118654752f));
                H[(size_t)mr * DFF_ + nc] = (_Float16)v;
            }
        }
}

// ---------------- GEMM2: out[scatter] = (H @ w2t^T + b2) * gate ----------------
__global__ __launch_bounds__(256) void gemm2_k(const _Float16* __restrict__ H,    // [B_*K_][DFF_]
                                               const _Float16* __restrict__ BwT,  // [D_][DFF_] (transposed w2)
                                               const float* __restrict__ bias,    // [D_]
                                               const int* __restrict__ sel_idx,   // [B_*K_]
                                               const float* __restrict__ gates,   // [B_*K_]
                                               float* __restrict__ out) {         // [B_][S_][D_]
    __shared__ _Float16 At[2][MT * LDT];
    __shared__ _Float16 Bl[2][NT * LDT];
    __shared__ int   s_tok[MT];
    __shared__ float s_gate[MT];
    const int tid  = threadIdx.x;
    const int n0   = blockIdx.x * NT;
    const int m0   = blockIdx.y * MT;
    const int lane = tid & 31;
    const int wv   = tid >> 5;
    const int lrow = lane & 15;
    const int half = lane >> 4;

    s_tok[tid]  = sel_idx[m0 + tid];
    s_gate[tid] = gates[m0 + tid];

    v8f acc[2][4];
#pragma unroll
    for (int s = 0; s < 2; ++s)
#pragma unroll
        for (int t = 0; t < 4; ++t)
#pragma unroll
            for (int e = 0; e < 8; ++e) acc[s][t][e] = 0.f;

    gemm_mainloop(H, DFF_, BwT, DFF_, m0, n0, DFF_,
                  (lds_h*)&At[0][0], (lds_h*)&At[1][0],
                  (lds_h*)&Bl[0][0], (lds_h*)&Bl[1][0],
                  tid, wv, lrow, half, acc);

#pragma unroll
    for (int s = 0; s < 2; ++s)
#pragma unroll
        for (int t = 0; t < 4; ++t) {
            int nc = n0 + t * 16 + lrow;
            float bv = bias[nc];
#pragma unroll
            for (int r = 0; r < 8; ++r) {
                int rl = wv * 32 + s * 16 + r + half * 8;
                int m  = m0 + rl;
                int b  = m >> 10;
                size_t orow = ((size_t)b * S_ + s_tok[rl]) * D_;
                out[orow + nc] = (acc[s][t][r] + bv) * s_gate[rl];
            }
        }
}

// ---------------- Host launcher ----------------
extern "C" void kernel_launch(void* const* d_in, const int* in_sizes, int n_in,
                              void* d_out, int out_size, void* d_ws, size_t ws_size,
                              hipStream_t stream) {
    const float* x  = (const float*)d_in[0];
    const float* rw = (const float*)d_in[1];
    const float* rb = (const float*)d_in[2];
    const float* w1 = (const float*)d_in[3];
    const float* b1 = (const float*)d_in[4];
    const float* w2 = (const float*)d_in[5];
    const float* b2 = (const float*)d_in[6];
    float* out = (float*)d_out;

    char* ws = (char*)d_ws;
    float*    scores  = (float*)(ws + 0);                       // 64 KB
    int*      sel_idx = (int*)(ws + 65536);                     // 16 KB
    float*    gates   = (float*)(ws + 81920);                   // 16 KB
    _Float16* w1t     = (_Float16*)(ws + 98304);                // 8 MB  [DFF][D]
    _Float16* w2t     = (_Float16*)(ws + 98304 + 8388608);      // 8 MB  [D][DFF]
    _Float16* selh    = (_Float16*)(ws + 98304 + 2 * 8388608);  // 8 MB  [B*K][D]
    _Float16* h16     = (_Float16*)(ws + 98304 + 3 * 8388608);  // 32 MB [B*K][DFF]

    zero_k<<<4096, 256, 0, stream>>>(out, (B_ * S_ * D_) / 4);
    router_k<<<(B_ * S_) / 8, 256, 0, stream>>>(x, rw, rb, scores);
    select_k<<<B_, 1024, 0, stream>>>(scores, sel_idx, gates);
    {   // w1 [D][DFF] -> w1t [DFF][D]
        dim3 blk(32, 8), grd(DFF_ / 32, D_ / 32);
        cvtT_k<<<grd, blk, 0, stream>>>(w1, w1t, D_, DFF_);
    }
    {   // w2 [DFF][D] -> w2t [D][DFF]
        dim3 blk(32, 8), grd(D_ / 32, DFF_ / 32);
        cvtT_k<<<grd, blk, 0, stream>>>(w2, w2t, DFF_, D_);
    }
    gather_k<<<B_ * K_, 256, 0, stream>>>(x, sel_idx, selh);
    {
        dim3 grid(DFF_ / NT, (B_ * K_) / MT);
        gemm1_k<<<grid, 256, 0, stream>>>(selh, w1t, b1, h16);
    }
    {
        dim3 grid(D_ / NT, (B_ * K_) / MT);
        gemm2_k<<<grid, 256, 0, stream>>>(h16, w2t, b2, sel_idx, gates, out);
    }
}
